// Attention_89189290868775
// MI455X (gfx1250) — matrix operand order repaired and checked by reading, not compiled, not run on hardware
//
#include <hip/hip_runtime.h>

typedef __attribute__((ext_vector_type(16))) __bf16 v16bf;
typedef __attribute__((ext_vector_type(8)))  float  v8f;
typedef __attribute__((ext_vector_type(4)))  unsigned v4u;
typedef __attribute__((ext_vector_type(8)))  int      v8i;
typedef __attribute__((ext_vector_type(4)))  int      v4i;

union BFrag { v16bf v; unsigned u[8]; };

__device__ __forceinline__ unsigned short f2bf(float f) {
    unsigned u = __float_as_uint(f);
    u += 0x7FFFu + ((u >> 16) & 1u);          // round-to-nearest-even
    return (unsigned short)(u >> 16);
}
__device__ __forceinline__ unsigned f2bf2(float lo, float hi) {
    return (unsigned)f2bf(lo) | ((unsigned)f2bf(hi) << 16);
}

__device__ __forceinline__ v8f wmma_bf16(v16bf a, v16bf b, v8f c) {
    return __builtin_amdgcn_wmma_f32_16x16x32_bf16(
        /*neg_a=*/false, a, /*neg_b=*/false, b,
        /*c_mod=*/(short)0, c, /*reuse_a=*/false, /*reuse_b=*/false);
}

#define NHEAD 12
#define HDIM  64
#define DIM   768
#define SEQ   4096
#define ROWS  8192          // B*N
#define QKSCALE 0.125f      // 64^-0.5

// ---------------------------------------------------------------------------
// TDM: async 2D bf16 tile load (64 rows x 64 cols) into LDS with +1-dword row
// padding (LDS row stride = 33 dwords). D# packing per CDNA5 ISA 8.3/8.4.
// 6-arg builtin variant (clang-23 / therock-10.0 headers).
// ---------------------------------------------------------------------------
__device__ __forceinline__ void tdm_load_tile64(unsigned lds_byte_off,
                                                const void* gaddr,
                                                unsigned row_stride_elems) {
    const unsigned long long ga = (unsigned long long)gaddr;
    v4u g0;
    g0[0] = 1u;                                   // count=1 (valid descriptor)
    g0[1] = lds_byte_off;                         // lds_addr
    g0[2] = (unsigned)ga;                         // global_addr[31:0]
    g0[3] = (unsigned)((ga >> 32) & 0x01FFFFFFu)  // global_addr[56:32]
          | (2u << 30);                           // type=2 (image)
    v8i g1;
    g1[0] = (int)((1u << 16)      // data_size = 2 bytes
                | (1u << 20)      // pad_enable
                | (4u << 22));    // pad_interval: every 32 dwords (one row)
                                  // pad_amount = 0 -> 1 dword pad
    g1[1] = (int)(64u << 16);     // tensor_dim0 = 64
    g1[2] = (int)(64u << 16);     // tensor_dim1 = 64
    g1[3] = (int)(64u << 16);     // tile_dim0 = 64
    g1[4] = (int)64u;             // tile_dim1 = 64, tile_dim2 = 0
    g1[5] = (int)row_stride_elems;// tensor_dim0_stride[31:0]
    g1[6] = 0;                    // stride hi / tensor_dim1_stride lo
    g1[7] = 0;
    v4i z4 = {0, 0, 0, 0};
    v8i z8 = {0, 0, 0, 0, 0, 0, 0, 0};
    __builtin_amdgcn_tensor_load_to_lds(g0, g1, z4, z4, z8, 0);
}

// ---------------------------------------------------------------------------
// Kernel 1: qkv = x @ W_qkv -> bf16 Q(scaled)/K in [B,H,N,64], V in [B,H,64,N]
// (V stored transposed so attention can TDM it directly).
// 256 threads (8 waves). Block tile 128x128, K-step 32.
// ---------------------------------------------------------------------------
__global__ __launch_bounds__(256) void qkv_gemm(
    const float* __restrict__ X, const float* __restrict__ W,
    unsigned short* __restrict__ Qw, unsigned short* __restrict__ Kw,
    unsigned short* __restrict__ Vw)
{
    __shared__ unsigned lA[128 * 17];   // bf16 pairs: A[m][k], stride 17 dwords
    __shared__ unsigned lB[128 * 17];   // bf16 pairs: B^T[n][k], stride 17 dwords

    const int tid  = threadIdx.x;
    const int row0 = blockIdx.y * 128;
    const int col0 = blockIdx.x * 128;
    const int wave = tid >> 5, lane = tid & 31;
    const int hh = lane >> 4, l16 = lane & 15;
    const int wm = (wave >> 1) * 32;    // wave M offset
    const int wn = (wave & 1) * 64;     // wave N offset

    v8f acc[2][4] = {};

    for (int kk = 0; kk < DIM; kk += 32) {
        // stage A tile (128x32 f32 -> bf16)
        {
            const int c4 = (tid & 7) * 4;
            const int rb = tid >> 3;
            for (int i = 0; i < 4; ++i) {
                const int m = rb + i * 32;
                float4 f = *(const float4*)(X + (size_t)(row0 + m) * DIM + kk + c4);
                lA[m * 17 + (c4 >> 1)]     = f2bf2(f.x, f.y);
                lA[m * 17 + (c4 >> 1) + 1] = f2bf2(f.z, f.w);
            }
        }
        // stage B tile transposed (32x128 f32 -> bf16 [n][k])
        {
            const int n  = tid & 127;
            const int kh = tid >> 7;     // 0..1
            float vals[16];
            for (int i = 0; i < 16; ++i)
                vals[i] = W[(size_t)(kk + kh * 16 + i) * (3 * DIM) + col0 + n];
            for (int i = 0; i < 8; ++i)
                lB[n * 17 + kh * 8 + i] = f2bf2(vals[2 * i], vals[2 * i + 1]);
        }
        __syncthreads();

        BFrag a[2], b[4];
        for (int mt = 0; mt < 2; ++mt)
            for (int p = 0; p < 8; ++p) {
                const int k = ((p & 4) ? 16 : 0) + 2 * (p & 3) + 8 * hh;
                a[mt].u[p] = lA[(wm + mt * 16 + l16) * 17 + (k >> 1)];
            }
        for (int nt = 0; nt < 4; ++nt)
            for (int v = 0; v < 8; ++v)
                b[nt].u[v] = lB[(wn + nt * 16 + l16) * 17 + ((v & 4) ? 8 : 0) + (v & 3) + 4 * hh];

        for (int mt = 0; mt < 2; ++mt)
            for (int nt = 0; nt < 4; ++nt)
                acc[mt][nt] = wmma_bf16(a[mt].v, b[nt].v, acc[mt][nt]);
        __syncthreads();
    }

    // epilogue: scatter bf16 into Q[B,H,N,64] (scaled), K[B,H,N,64], V^T[B,H,64,N]
    for (int mt = 0; mt < 2; ++mt)
        for (int nt = 0; nt < 4; ++nt)
            for (int r = 0; r < 8; ++r) {
                const int grow = row0 + wm + mt * 16 + r + 8 * hh;
                const int gcol = col0 + wn + nt * 16 + l16;
                const int seg  = gcol / DIM;
                const int cc   = gcol - seg * DIM;
                const int h    = cc >> 6, d = cc & 63;
                const int bb   = grow >> 12, n = grow & 4095;
                const float val = acc[mt][nt][r];
                const size_t head = (size_t)bb * NHEAD + h;
                if (seg == 0)
                    Qw[(head * SEQ + n) * HDIM + d] = f2bf(val * QKSCALE);
                else if (seg == 1)
                    Kw[(head * SEQ + n) * HDIM + d] = f2bf(val);
                else
                    Vw[(head * HDIM + d) * SEQ + n] = f2bf(val);
            }
}

// ---------------------------------------------------------------------------
// Kernel 2: flash attention with double-buffered TDM staging of K/V tiles.
// Grid (64 q-blocks, 24 b*h). 128 threads (4 waves), 16 query rows per wave.
// ---------------------------------------------------------------------------
__global__ __launch_bounds__(128) void fa_kernel(
    const unsigned short* __restrict__ Qg, const unsigned short* __restrict__ Kg,
    const unsigned short* __restrict__ Vg, float* __restrict__ O)
{
    __shared__ unsigned lK[2][64 * 33];      // K tile [key][d], pairs along d
    __shared__ unsigned lV[2][64 * 33];      // V tile [d][key], pairs along key
    __shared__ unsigned lP[4 * 16 * 33];     // per-wave P [m][key]

    const int tid  = threadIdx.x;
    const int wave = tid >> 5, lane = tid & 31;
    const int hh = lane >> 4, l16 = lane & 15;
    const int qb = blockIdx.x;               // 0..63
    const int bh = blockIdx.y;               // 0..23
    const int b  = bh / NHEAD, h = bh - b * NHEAD;
    const int NIT = SEQ / 64;

    const unsigned short* Kh = Kg + (size_t)bh * SEQ * HDIM;   // [4096][64]
    const unsigned short* Vh = Vg + (size_t)bh * HDIM * SEQ;   // [64][4096]

    // Q fragments: straight from global (row-major bf16, pairs contiguous)
    BFrag qf[2];
    {
        const unsigned short* qrow =
            Qg + ((size_t)bh * SEQ + qb * 64 + wave * 16 + l16) * HDIM;
        for (int c = 0; c < 2; ++c)
            for (int p = 0; p < 8; ++p) {
                const int k = c * 32 + ((p & 4) ? 16 : 0) + 2 * (p & 3) + 8 * hh;
                qf[c].u[p] = *(const unsigned*)(qrow + k);
            }
    }

    float rm[8], rl[8];
    v8f o[4] = {};
    for (int r = 0; r < 8; ++r) { rm[r] = -1e30f; rl[r] = 0.f; }

    // preload tile 0 (wave 0 drives the Tensor Data Mover)
    if (wave == 0) {
        tdm_load_tile64((unsigned)(unsigned long long)(void*)&lK[0][0],
                        Kh, HDIM);                       // K rows: stride 64
        tdm_load_tile64((unsigned)(unsigned long long)(void*)&lV[0][0],
                        Vh, SEQ);                        // V^T rows: stride 4096
    }

    for (int jb = 0; jb < NIT; ++jb) {
        const int cur = jb & 1;
        if (wave == 0) {
            if (jb + 1 < NIT) {   // kick off next tile while we compute
                tdm_load_tile64((unsigned)(unsigned long long)(void*)&lK[cur ^ 1][0],
                                Kh + (size_t)(jb + 1) * 64 * HDIM, HDIM);
                tdm_load_tile64((unsigned)(unsigned long long)(void*)&lV[cur ^ 1][0],
                                Vh + (size_t)(jb + 1) * 64, SEQ);
                __builtin_amdgcn_s_wait_tensorcnt(2);  // current pair retired
            } else {
                __builtin_amdgcn_s_wait_tensorcnt(0);
            }
        }
        __syncthreads();
        const unsigned* Kt = lK[cur];
        const unsigned* Vt = lV[cur];

        // S = Q K^T  (16 x 64, already scaled via Q)
        v8f s[4];
        for (int nt = 0; nt < 4; ++nt) {
            BFrag kb0, kb1;
            for (int v = 0; v < 8; ++v) {
                kb0.u[v] = Kt[(nt * 16 + l16) * 33 + ((v & 4) ? 8 : 0) + (v & 3) + 4 * hh];
                kb1.u[v] = Kt[(nt * 16 + l16) * 33 + 16 + ((v & 4) ? 8 : 0) + (v & 3) + 4 * hh];
            }
            v8f t = {};
            t = wmma_bf16(qf[0].v, kb0.v, t);
            t = wmma_bf16(qf[1].v, kb1.v, t);
            s[nt] = t;
        }

        // online softmax (f32)
        float alpha[8];
        for (int r = 0; r < 8; ++r) {
            float mx = fmaxf(fmaxf(s[0][r], s[1][r]), fmaxf(s[2][r], s[3][r]));
            mx = fmaxf(mx, __shfl_xor(mx, 1, 32));
            mx = fmaxf(mx, __shfl_xor(mx, 2, 32));
            mx = fmaxf(mx, __shfl_xor(mx, 4, 32));
            mx = fmaxf(mx, __shfl_xor(mx, 8, 32));
            const float nm = fmaxf(rm[r], mx);
            alpha[r] = __expf(rm[r] - nm);
            rm[r] = nm;
            float rs = 0.f;
            for (int nt = 0; nt < 4; ++nt) {
                const float p = __expf(s[nt][r] - nm);
                s[nt][r] = p;
                rs += p;
            }
            rs += __shfl_xor(rs, 1, 32);
            rs += __shfl_xor(rs, 2, 32);
            rs += __shfl_xor(rs, 4, 32);
            rs += __shfl_xor(rs, 8, 32);
            rl[r] = rl[r] * alpha[r] + rs;
        }
        for (int nt = 0; nt < 4; ++nt)
            for (int r = 0; r < 8; ++r)
                o[nt][r] *= alpha[r];

        // P: C-layout -> A-layout through per-wave LDS (bf16)
        {
            unsigned short* lP16 = (unsigned short*)lP;
            for (int nt = 0; nt < 4; ++nt)
                for (int r = 0; r < 8; ++r)
                    lP16[(wave * 16 + r + 8 * hh) * 66 + nt * 16 + l16] =
                        f2bf(s[nt][r]);
        }
        BFrag pf[2];
        for (int c = 0; c < 2; ++c)
            for (int p = 0; p < 8; ++p) {
                const int k = c * 32 + ((p & 4) ? 16 : 0) + 2 * (p & 3) + 8 * hh;
                pf[c].u[p] = lP[(wave * 16 + l16) * 33 + (k >> 1)];
            }

        // O += P V
        for (int nt = 0; nt < 4; ++nt) {
            BFrag vb0, vb1;
            for (int v = 0; v < 8; ++v) {
                vb0.u[v] = Vt[(nt * 16 + l16) * 33 + ((v & 4) ? 8 : 0) + (v & 3) + 4 * hh];
                vb1.u[v] = Vt[(nt * 16 + l16) * 33 + 16 + ((v & 4) ? 8 : 0) + (v & 3) + 4 * hh];
            }
            o[nt] = wmma_bf16(pf[0].v, vb0.v, o[nt]);
            o[nt] = wmma_bf16(pf[1].v, vb1.v, o[nt]);
        }
        __syncthreads();   // everyone done with this buffer before TDM reuses it
    }

    // epilogue: O /= l, store f32 into [B, N, H*64]
    for (int nt = 0; nt < 4; ++nt)
        for (int r = 0; r < 8; ++r) {
            const int m = r + 8 * hh;
            const int n = qb * 64 + wave * 16 + m;
            const float inv = 1.0f / rl[r];
            O[((size_t)b * SEQ + n) * DIM + h * HDIM + nt * 16 + l16] =
                o[nt][r] * inv;
        }
}

// ---------------------------------------------------------------------------
// Kernel 3: out = O @ W_out + b_out (f32 out). Same GEMM core as kernel 1.
// ---------------------------------------------------------------------------
__global__ __launch_bounds__(256) void out_gemm(
    const float* __restrict__ X, const float* __restrict__ W,
    const float* __restrict__ bias, float* __restrict__ out)
{
    __shared__ unsigned lA[128 * 17];
    __shared__ unsigned lB[128 * 17];

    const int tid  = threadIdx.x;
    const int row0 = blockIdx.y * 128;
    const int col0 = blockIdx.x * 128;
    const int wave = tid >> 5, lane = tid & 31;
    const int hh = lane >> 4, l16 = lane & 15;
    const int wm = (wave >> 1) * 32;
    const int wn = (wave & 1) * 64;

    v8f acc[2][4] = {};

    for (int kk = 0; kk < DIM; kk += 32) {
        {
            const int c4 = (tid & 7) * 4;
            const int rb = tid >> 3;
            for (int i = 0; i < 4; ++i) {
                const int m = rb + i * 32;
                float4 f = *(const float4*)(X + (size_t)(row0 + m) * DIM + kk + c4);
                lA[m * 17 + (c4 >> 1)]     = f2bf2(f.x, f.y);
                lA[m * 17 + (c4 >> 1) + 1] = f2bf2(f.z, f.w);
            }
        }
        {
            const int n  = tid & 127;
            const int kh = tid >> 7;
            float vals[16];
            for (int i = 0; i < 16; ++i)
                vals[i] = W[(size_t)(kk + kh * 16 + i) * DIM + col0 + n];
            for (int i = 0; i < 8; ++i)
                lB[n * 17 + kh * 8 + i] = f2bf2(vals[2 * i], vals[2 * i + 1]);
        }
        __syncthreads();

        BFrag a[2], b[4];
        for (int mt = 0; mt < 2; ++mt)
            for (int p = 0; p < 8; ++p) {
                const int k = ((p & 4) ? 16 : 0) + 2 * (p & 3) + 8 * hh;
                a[mt].u[p] = lA[(wm + mt * 16 + l16) * 17 + (k >> 1)];
            }
        for (int nt = 0; nt < 4; ++nt)
            for (int v = 0; v < 8; ++v)
                b[nt].u[v] = lB[(wn + nt * 16 + l16) * 17 + ((v & 4) ? 8 : 0) + (v & 3) + 4 * hh];

        for (int mt = 0; mt < 2; ++mt)
            for (int nt = 0; nt < 4; ++nt)
                acc[mt][nt] = wmma_bf16(a[mt].v, b[nt].v, acc[mt][nt]);
        __syncthreads();
    }

    for (int mt = 0; mt < 2; ++mt)
        for (int nt = 0; nt < 4; ++nt)
            for (int r = 0; r < 8; ++r) {
                const int grow = row0 + wm + mt * 16 + r + 8 * hh;
                const int gcol = col0 + wn + nt * 16 + l16;
                out[(size_t)grow * DIM + gcol] = acc[mt][nt][r] + bias[gcol];
            }
}

// ---------------------------------------------------------------------------
extern "C" void kernel_launch(void* const* d_in, const int* in_sizes, int n_in,
                              void* d_out, int out_size, void* d_ws, size_t ws_size,
                              hipStream_t stream) {
    const float* x    = (const float*)d_in[0];   // [2,4096,768]
    const float* Wqkv = (const float*)d_in[1];   // [768,2304]
    const float* Wout = (const float*)d_in[2];   // [768,768]
    const float* bout = (const float*)d_in[3];   // [768]
    float* out = (float*)d_out;                  // [2,4096,768]

    const size_t headElems = (size_t)2 * NHEAD * SEQ * HDIM;  // 6.29M
    unsigned short* Qw = (unsigned short*)d_ws;
    unsigned short* Kw = Qw + headElems;
    unsigned short* Vw = Kw + headElems;
    float* Ow = (float*)(Vw + headElems);        // [8192,768] f32

    dim3 g1(3 * DIM / 128, ROWS / 128);          // (18, 64)
    qkv_gemm<<<g1, 256, 0, stream>>>(x, Wqkv, Qw, Kw, Vw);

    dim3 g2(SEQ / 64, 2 * NHEAD);                // (64, 24)
    fa_kernel<<<g2, 128, 0, stream>>>(Qw, Kw, Vw, Ow);

    dim3 g3(DIM / 128, ROWS / 128);              // (6, 64)
    out_gemm<<<g3, 256, 0, stream>>>(Ow, Wout, bout, out);
}
